// EFLSTM_68693706932301
// MI455X (gfx1250) — matrix operand order repaired and checked by reading, not compiled, not run on hardware
//
#include <hip/hip_runtime.h>

#define DEV __device__ __forceinline__

typedef __attribute__((ext_vector_type(16))) __bf16        v16bf;
typedef __attribute__((ext_vector_type(8)))  float          v8f;
typedef __attribute__((ext_vector_type(8)))  unsigned short su8;

union FragU {
  v16bf v;
  struct { su8 lo; su8 hi; } p;
};

// ---- helpers -------------------------------------------------------------

DEV unsigned short f2bf(float f) {
  union { float f; unsigned u; } v; v.f = f;
  unsigned r = v.u + 0x7FFFu + ((v.u >> 16) & 1u);   // round-to-nearest-even
  return (unsigned short)(r >> 16);
}

// A-matrix fragment (16x32 bf16): lane holds row = lane&15,
// contiguous K = kb..kb+15, kb = 16*(lane>=16).  Two b128 loads.
DEV v16bf ldA(const unsigned short* __restrict__ row, int k) {
  FragU f;
  f.p.lo = *reinterpret_cast<const su8*>(row + k);
  f.p.hi = *reinterpret_cast<const su8*>(row + k + 8);
  return f.v;
}
// B-matrix fragment (32x16 bf16): lane holds col = lane&15,
// contiguous K = kb..kb+15, kb = 16*(lane>=16).  Two b128 loads.
DEV v16bf ldB(const unsigned short* __restrict__ col, int k) {
  FragU f;
  f.p.lo = *reinterpret_cast<const su8*>(col + k);
  f.p.hi = *reinterpret_cast<const su8*>(col + k + 8);
  return f.v;
}

DEV v8f wmma_bf(v16bf a, v16bf b, v8f c) {
  return __builtin_amdgcn_wmma_f32_16x16x32_bf16(
      /*neg_a=*/false, a, /*neg_b=*/false, b,
      /*c_mod=*/(short)0, c, /*reuse_a=*/false, /*reuse_b=*/false);
}

DEV float sigm(float x) { return 1.0f / (1.0f + __expf(-x)); }

// ---- prep kernels --------------------------------------------------------

// wcat[g, 0:512] = W_ih[g,:], wcat[g, 512:1536] = W_hh[g,:]   (bf16)
__global__ void cvt_wcat_kernel(const float* __restrict__ wih,
                                const float* __restrict__ whh,
                                unsigned short* __restrict__ wcat) {
  int i = blockIdx.x * blockDim.x + threadIdx.x;       // 0 .. 4096*1536-1
  int g = i / 1536, k = i - g * 1536;
  float v = (k < 512) ? wih[(size_t)g * 512 + k] : whh[(size_t)g * 1024 + (k - 512)];
  wcat[i] = f2bf(v);
}

__global__ void cvt_bf_kernel(const float* __restrict__ src,
                              unsigned short* __restrict__ dst, int n) {
  int i = blockIdx.x * blockDim.x + threadIdx.x;
  if (i < n) dst[i] = f2bf(src[i]);
}

__global__ void init_state_kernel(const float* __restrict__ bih,
                                  const float* __restrict__ bhh,
                                  float* __restrict__ bsum,
                                  unsigned short* __restrict__ h0,
                                  float* __restrict__ c) {
  int i = blockIdx.x * blockDim.x + threadIdx.x;       // 0 .. 262143
  h0[i] = 0;
  c[i]  = 0.0f;
  if (i < 4096) bsum[i] = bih[i] + bhh[i];
}

// ---- fused LSTM step: gates GEMM (WMMA) + cell ---------------------------
// grid (32, 8): blockIdx.x = 32-col tile of H, blockIdx.y = 32-row tile of N.
// Wave (mi, gate) computes a 16x32 tile of gate `gate` via 2 accumulators.

__global__ __launch_bounds__(256) void lstm_step_kernel(
    const unsigned short* __restrict__ xbf,   // [256,128,512] bf16
    const unsigned short* __restrict__ wcat,  // [4096,1536]   bf16
    const float*          __restrict__ bsum,  // [4096]
    const unsigned short* __restrict__ hin,   // [256,1024]    bf16
    unsigned short*       __restrict__ hout,  // [256,1024]    bf16
    float*                __restrict__ cst,   // [256,1024]    f32
    int t) {
  __shared__ float lds[4][32][33];

  const int tid   = threadIdx.x;
  const int lane  = tid & 31;
  const int wave  = tid >> 5;
  const int mi    = wave & 1;
  const int gate  = wave >> 1;
  const int hblk  = blockIdx.x;
  const int mblk  = blockIdx.y;
  const int rlane = lane & 15;
  const int hi    = lane >> 4;
  const int kA    = hi << 3;        // 0 or 8
  const int kB    = hi << 4;        // 0 or 16
  const int row   = mblk * 32 + mi * 16 + rlane;

  v8f acc0 = {}; v8f acc1 = {};

  // Phase 1: x(t) contribution (K = 0..511 of concatenated contraction)
  {
    const unsigned short* a  = xbf + (size_t)row * (128 * 512) + (size_t)t * 512;
    const unsigned short* b0 = wcat + (size_t)(gate * 1024 + hblk * 32 + rlane) * 1536;
    const unsigned short* b1 = b0 + (size_t)16 * 1536;
#pragma unroll 4
    for (int kc = 0; kc < 16; ++kc) {
      const int ka = kc * 32 + kB, kb = kc * 32 + kB;
      v16bf af = ldA(a, ka);
      acc0 = wmma_bf(af, ldB(b0, kb), acc0);
      acc1 = wmma_bf(af, ldB(b1, kb), acc1);
    }
  }
  // Phase 2: h(t-1) contribution (K = 512..1535)
  {
    const unsigned short* a  = hin + (size_t)row * 1024;
    const unsigned short* b0 = wcat + (size_t)(gate * 1024 + hblk * 32 + rlane) * 1536 + 512;
    const unsigned short* b1 = b0 + (size_t)16 * 1536;
#pragma unroll 4
    for (int kc = 0; kc < 32; ++kc) {
      const int ka = kc * 32 + kB, kb = kc * 32 + kB;
      v16bf af = ldA(a, ka);
      acc0 = wmma_bf(af, ldB(b0, kb), acc0);
      acc1 = wmma_bf(af, ldB(b1, kb), acc1);
    }
  }

  // C/D layout: VGPR v -> m = 8*(lane>=16)+v, n = lane&15
  const int r0 = mi * 16 + (hi << 3);
#pragma unroll
  for (int v = 0; v < 8; ++v) {
    lds[gate][r0 + v][rlane]      = acc0[v];
    lds[gate][r0 + v][16 + rlane] = acc1[v];
  }
  __syncthreads();

  // Fused LSTM cell: 32x32 tile, 4 elements/thread
#pragma unroll
  for (int e = 0; e < 4; ++e) {
    int idx  = tid + e * 256;
    int r    = idx >> 5;
    int cc   = idx & 31;
    int hcol = hblk * 32 + cc;
    int R    = mblk * 32 + r;
    float gi = lds[0][r][cc] + bsum[hcol];
    float gf = lds[1][r][cc] + bsum[1024 + hcol];
    float gg = lds[2][r][cc] + bsum[2048 + hcol];
    float go = lds[3][r][cc] + bsum[3072 + hcol];
    float iv = sigm(gi), fv = sigm(gf), ov = sigm(go);
    float gv = tanhf(gg);
    size_t ci = (size_t)R * 1024 + hcol;
    float cn = fv * cst[ci] + iv * gv;
    cst[ci]  = cn;
    hout[ci] = f2bf(ov * tanhf(cn));
  }
}

// ---- tail MLP 1: pre_output = relu(h @ W1^T + b1) ------------------------
// grid (4, 8); wave (mi, cg) -> 16 rows x 64 cols (4 accumulators)

__global__ __launch_bounds__(256) void mlp1_kernel(
    const unsigned short* __restrict__ hb,     // [256,1024] bf16
    const unsigned short* __restrict__ w1b,    // [1024,1024] bf16
    const float*          __restrict__ b1,
    float*                __restrict__ preout, // [256,1024] f32 (d_out part 2)
    unsigned short*       __restrict__ preb) { // [256,1024] bf16
  const int tid = threadIdx.x, lane = tid & 31, wave = tid >> 5;
  const int mi = wave & 1, cg = wave >> 1;
  const int rlane = lane & 15, hi = lane >> 4;
  const int kA = hi << 3, kB = hi << 4;
  const int row = blockIdx.y * 32 + mi * 16 + rlane;
  const int colbase = blockIdx.x * 256 + cg * 64;

  v8f acc[4] = {};
  const unsigned short* a = hb + (size_t)row * 1024;
  const unsigned short* bp[4];
#pragma unroll
  for (int s = 0; s < 4; ++s) bp[s] = w1b + (size_t)(colbase + s * 16 + rlane) * 1024;

#pragma unroll 2
  for (int kc = 0; kc < 32; ++kc) {
    const int ka = kc * 32 + kB, kb = kc * 32 + kB;
    v16bf af = ldA(a, ka);
#pragma unroll
    for (int s = 0; s < 4; ++s) acc[s] = wmma_bf(af, ldB(bp[s], kb), acc[s]);
  }

  const int erow = blockIdx.y * 32 + mi * 16 + (hi << 3);
#pragma unroll
  for (int s = 0; s < 4; ++s) {
    int col = colbase + s * 16 + rlane;
    float bias = b1[col];
#pragma unroll
    for (int v = 0; v < 8; ++v) {
      float val = acc[s][v] + bias;
      val = val > 0.0f ? val : 0.0f;
      size_t o = (size_t)(erow + v) * 1024 + col;
      preout[o] = val;
      preb[o]   = f2bf(val);
    }
  }
}

// ---- tail MLP 2: output = pre_output @ W2^T + b2 -------------------------
// grid (8); wave (mi, cg) -> 16 rows x 32 cols (2 accumulators)

__global__ __launch_bounds__(256) void mlp2_kernel(
    const unsigned short* __restrict__ preb,  // [256,1024] bf16
    const unsigned short* __restrict__ w2b,   // [128,1024] bf16
    const float*          __restrict__ b2,
    float*                __restrict__ out) { // [256,128] f32 (d_out part 1)
  const int tid = threadIdx.x, lane = tid & 31, wave = tid >> 5;
  const int mi = wave & 1, cg = wave >> 1;
  const int rlane = lane & 15, hi = lane >> 4;
  const int kA = hi << 3, kB = hi << 4;
  const int row = blockIdx.x * 32 + mi * 16 + rlane;
  const int colbase = cg * 32;

  v8f acc0 = {}; v8f acc1 = {};
  const unsigned short* a  = preb + (size_t)row * 1024;
  const unsigned short* b0 = w2b + (size_t)(colbase + rlane) * 1024;
  const unsigned short* b1 = b0 + (size_t)16 * 1024;
#pragma unroll 4
  for (int kc = 0; kc < 32; ++kc) {
    const int ka = kc * 32 + kB, kb = kc * 32 + kB;
    v16bf af = ldA(a, ka);
    acc0 = wmma_bf(af, ldB(b0, kb), acc0);
    acc1 = wmma_bf(af, ldB(b1, kb), acc1);
  }
  const int erow = blockIdx.x * 32 + mi * 16 + (hi << 3);
#pragma unroll
  for (int v = 0; v < 8; ++v) {
    out[(size_t)(erow + v) * 128 + colbase + rlane]      = acc0[v] + b2[colbase + rlane];
    out[(size_t)(erow + v) * 128 + colbase + 16 + rlane] = acc1[v] + b2[colbase + 16 + rlane];
  }
}

// ---- host ----------------------------------------------------------------

extern "C" void kernel_launch(void* const* d_in, const int* in_sizes, int n_in,
                              void* d_out, int out_size, void* d_ws, size_t ws_size,
                              hipStream_t stream) {
  const float* x    = (const float*)d_in[0];  // [256,128,512]
  const float* W_ih = (const float*)d_in[1];  // [4096,512]
  const float* W_hh = (const float*)d_in[2];  // [4096,1024]
  const float* b_ih = (const float*)d_in[3];  // [4096]
  const float* b_hh = (const float*)d_in[4];  // [4096]
  const float* W1   = (const float*)d_in[5];  // [1024,1024]
  const float* b1   = (const float*)d_in[6];  // [1024]
  const float* W2   = (const float*)d_in[7];  // [128,1024]
  const float* b2   = (const float*)d_in[8];  // [128]

  float* out    = (float*)d_out;              // [256,128]
  float* preout = out + 256 * 128;            // [256,1024]

  char* p = (char*)d_ws;
  unsigned short* wcat = (unsigned short*)p;  p += (size_t)4096 * 1536 * 2;
  unsigned short* xbf  = (unsigned short*)p;  p += (size_t)256 * 128 * 512 * 2;
  unsigned short* w1b  = (unsigned short*)p;  p += (size_t)1024 * 1024 * 2;
  unsigned short* w2b  = (unsigned short*)p;  p += (size_t)128 * 1024 * 2;
  float*          bsum = (float*)p;           p += (size_t)4096 * 4;
  unsigned short* hb0  = (unsigned short*)p;  p += (size_t)256 * 1024 * 2;
  unsigned short* hb1  = (unsigned short*)p;  p += (size_t)256 * 1024 * 2;
  float*          cst  = (float*)p;           p += (size_t)256 * 1024 * 4;
  unsigned short* preb = (unsigned short*)p;  p += (size_t)256 * 1024 * 2;

  cvt_wcat_kernel<<<(4096 * 1536) / 256, 256, 0, stream>>>(W_ih, W_hh, wcat);
  cvt_bf_kernel<<<(256 * 128 * 512) / 256, 256, 0, stream>>>(x, xbf, 256 * 128 * 512);
  cvt_bf_kernel<<<(1024 * 1024) / 256, 256, 0, stream>>>(W1, w1b, 1024 * 1024);
  cvt_bf_kernel<<<(128 * 1024) / 256, 256, 0, stream>>>(W2, w2b, 128 * 1024);
  init_state_kernel<<<(256 * 1024) / 256, 256, 0, stream>>>(b_ih, b_hh, bsum, hb0, cst);

  for (int t = 0; t < 128; ++t) {
    const unsigned short* hin = (t & 1) ? hb1 : hb0;
    unsigned short*      hout = (t & 1) ? hb0 : hb1;
    lstm_step_kernel<<<dim3(32, 8), 256, 0, stream>>>(xbf, wcat, bsum, hin, hout, cst, t);
  }
  // after 128 steps the freshest h is in hb0
  mlp1_kernel<<<dim3(4, 8), 256, 0, stream>>>(hb0, w1b, b1, preout, preb);
  mlp2_kernel<<<8, 256, 0, stream>>>(preb, w2b, b2, out);
}
